// EntropyOptimizedMLP_10720238370984
// MI455X (gfx1250) — compile-verified, matrix operand order checked by reading, not run on hardware
//
#include <hip/hip_runtime.h>
#include <hip/hip_bf16.h>
#include <math.h>

// ---------------------------------------------------------------------------
// EntropyOptimizedMLP for MI455X (gfx1250, wave32, WMMA)
//   B=8192, IN=4096, H1=1024, H2=512, OUT=10
// ---------------------------------------------------------------------------

#define BATCH 8192
#define IN_F  4096
#define H1_F  1024
#define H2_F  512
#define OUT_F 10

typedef _Float16 v16h __attribute__((ext_vector_type(16)));
typedef float    v8f  __attribute__((ext_vector_type(8)));
typedef float    v2f  __attribute__((ext_vector_type(2)));

#if __has_builtin(__builtin_amdgcn_wmma_f32_16x16x4_f32)
#define HAVE_WMMA_F32 1
#else
#define HAVE_WMMA_F32 0
#endif

// ---------------------------------------------------------------------------
// Per-sample 256-bin histogram entropy, accumulated (normalized) into *acc.
// One block (256 threads) per sample. LDS reductions + LDS histogram atomics.
// ---------------------------------------------------------------------------
__global__ __launch_bounds__(256)
void entropy_kernel(const float* __restrict__ X, int F, float inv_scale,
                    float* __restrict__ acc)
{
    __shared__ float red[256];
    __shared__ int   hist[256];

    const int b = blockIdx.x;
    const int t = threadIdx.x;
    const float* x = X + (size_t)b * F;

    float vmin =  3.402823e38f;
    float vmax = -3.402823e38f;
    for (int i = t; i < F; i += 256) {
        float v = x[i];
        vmin = fminf(vmin, v);
        vmax = fmaxf(vmax, v);
    }
    red[t] = vmin; __syncthreads();
    for (int s = 128; s > 0; s >>= 1) {
        if (t < s) red[t] = fminf(red[t], red[t + s]);
        __syncthreads();
    }
    const float xmin = red[0];
    __syncthreads();
    red[t] = vmax; __syncthreads();
    for (int s = 128; s > 0; s >>= 1) {
        if (t < s) red[t] = fmaxf(red[t], red[t + s]);
        __syncthreads();
    }
    const float xmax = red[0];
    __syncthreads();

    const float rng = fmaxf(xmax - xmin, 1e-12f);
    const float scl = 256.0f / rng;
    hist[t] = 0;
    __syncthreads();
    for (int i = t; i < F; i += 256) {
        int idx = (int)((x[i] - xmin) * scl);
        idx = idx < 0 ? 0 : (idx > 255 ? 255 : idx);
        atomicAdd(&hist[idx], 1);
    }
    __syncthreads();

    const int c = hist[t];
    float contrib = 0.0f;
    if (c > 0) {
        float p = (float)c / (float)F;
        contrib = -p * log2f(p);
    }
    red[t] = contrib;
    __syncthreads();
    for (int s = 128; s > 0; s >>= 1) {
        if (t < s) red[t] += red[t + s];
        __syncthreads();
    }
    if (t == 0) atomicAdd(acc, red[0] * inv_scale);  // inv_scale = 1/(8*BATCH)
}

__global__ void zero_acc_kernel(float* acc)
{
    acc[0] = 0.0f; acc[1] = 0.0f; acc[2] = 0.0f;
}

// ---------------------------------------------------------------------------
// Fragment loaders (per-lane contiguous vector loads, EXEC untouched).
// ---------------------------------------------------------------------------
__device__ __forceinline__ v16h load_f16_frag(const float* __restrict__ p)
{
    // K-halves {0..7} and {16..23} of the lane's 32-wide slab (hi-lane shift
    // already folded into p by the caller) -> four b128 loads + cvt.
    const float4 x0 = *(const float4*)(p);
    const float4 x1 = *(const float4*)(p + 4);
    const float4 x2 = *(const float4*)(p + 16);
    const float4 x3 = *(const float4*)(p + 20);
    v16h r;
    r[0]=(_Float16)x0.x;  r[1]=(_Float16)x0.y;  r[2]=(_Float16)x0.z;  r[3]=(_Float16)x0.w;
    r[4]=(_Float16)x1.x;  r[5]=(_Float16)x1.y;  r[6]=(_Float16)x1.z;  r[7]=(_Float16)x1.w;
    r[8]=(_Float16)x2.x;  r[9]=(_Float16)x2.y;  r[10]=(_Float16)x2.z; r[11]=(_Float16)x2.w;
    r[12]=(_Float16)x3.x; r[13]=(_Float16)x3.y; r[14]=(_Float16)x3.z; r[15]=(_Float16)x3.w;
    return r;
}

__device__ __forceinline__ v2f load_f32_frag(const float* __restrict__ p)
{
    const float2 x = *(const float2*)(p);
    v2f r; r[0] = x.x; r[1] = x.y;
    return r;
}

// ---------------------------------------------------------------------------
// C[M,N] = A[M,K] * W[N,K]^T + bias, optional ReLU, precision chosen at
// runtime from *scale_mean (uniform scalar branch; EXEC all-ones at WMMAs):
//   mean < 0.5  -> fp16 inputs, V_WMMA_F32_16X16X32_F16
//   else        -> fp32 inputs, V_WMMA_F32_16X16X4_F32
//
// 256 threads = 8 wave32 arranged 2(M) x 4(N); each wave register-blocks a
// 32x32 output (2x2 WMMA tiles, 4 v8f accumulators) so each k-step shares
// 2 A-fragments across 2 B-fragments: 4 WMMAs per 4 fragment loads (2x the
// arithmetic intensity of 1 tile/wave). Block tile = 64 x 128.
//
// M (=8192) is a multiple of 64, so row loads are never out of range.
// Out-of-range columns clamp their W row for loads and are masked at stores.
//
// Operand layouts per CDNA5 ISA 7.12.2 (wave32):
//   A f16 16x32: lane l in [0,16): M=l; half j holds K=(j<8?j:j+8)+hi*8
//   B mirrors A with N in place of M (W is [N,K] row-major).
//   A f32 16x4:  elem e holds K = e + hi*2
//   C/D f32: lane l -> N=l&15; VGPR r -> M = r + hi*8
// ---------------------------------------------------------------------------
__global__ __launch_bounds__(256)
void gemm_entropy_branch(const float* __restrict__ A, const float* __restrict__ W,
                         const float* __restrict__ bias, float* __restrict__ C,
                         const float* __restrict__ scale_mean,
                         int M, int N, int K, int relu)
{
    const int lane = threadIdx.x & 31;
    const int wave = threadIdx.x >> 5;
    const int l    = lane & 15;
    const int hi   = lane >> 4;        // 0 = lanes 0-15, 1 = lanes 16-31
    const int wm   = wave & 1;         // 2 waves along M
    const int wn   = wave >> 1;        // 4 waves along N

    const int m0 = blockIdx.x * 64  + wm * 32;   // wave 32-row base
    const int n0 = blockIdx.y * 128 + wn * 32;   // wave 32-col base

    const int row0 = m0 + l;                     // A rows (always < M)
    const int row1 = m0 + 16 + l;
    const int c0   = n0 + l;                     // C columns
    const int c1   = n0 + 16 + l;
    const int c0L  = c0 < N ? c0 : (N - 1);      // clamped for loads
    const int c1L  = c1 < N ? c1 : (N - 1);

    const float* __restrict__ A0 = A + (size_t)row0 * K;
    const float* __restrict__ A1 = A + (size_t)row1 * K;
    const float* __restrict__ B0 = W + (size_t)c0L * K;
    const float* __restrict__ B1 = W + (size_t)c1L * K;

    // accumulators preloaded with bias (broadcast along M)
    v8f acc00, acc01, acc10, acc11;
    const float bv0 = bias[c0L];
    const float bv1 = bias[c1L];
#pragma unroll
    for (int r = 0; r < 8; ++r) {
        acc00[r] = bv0; acc10[r] = bv0;
        acc01[r] = bv1; acc11[r] = bv1;
    }

    const bool use_fp16 = (*scale_mean) < 0.5f;   // uniform scalar branch

    if (use_fp16) {
        const int off = hi ? 8 : 0;
        for (int k = 0; k < K; k += 32) {
            const v16h aF0 = load_f16_frag(A0 + off + k);
            const v16h aF1 = load_f16_frag(A1 + off + k);
            const v16h bF0 = load_f16_frag(B0 + off + k);
            const v16h bF1 = load_f16_frag(B1 + off + k);
            acc00 = __builtin_amdgcn_wmma_f32_16x16x32_f16(false, aF0, false, bF0, (short)0, acc00, false, false);
            acc01 = __builtin_amdgcn_wmma_f32_16x16x32_f16(false, aF0, false, bF1, (short)0, acc01, false, false);
            acc10 = __builtin_amdgcn_wmma_f32_16x16x32_f16(false, aF1, false, bF0, (short)0, acc10, false, false);
            acc11 = __builtin_amdgcn_wmma_f32_16x16x32_f16(false, aF1, false, bF1, (short)0, acc11, false, false);
        }
    } else {
#if HAVE_WMMA_F32
        const int off = hi ? 2 : 0;
#pragma unroll 2
        for (int k = 0; k < K; k += 4) {
            const v2f aF0 = load_f32_frag(A0 + off + k);
            const v2f aF1 = load_f32_frag(A1 + off + k);
            const v2f bF0 = load_f32_frag(B0 + off + k);
            const v2f bF1 = load_f32_frag(B1 + off + k);
            acc00 = __builtin_amdgcn_wmma_f32_16x16x4_f32(false, aF0, false, bF0, (short)0, acc00, false, false);
            acc01 = __builtin_amdgcn_wmma_f32_16x16x4_f32(false, aF0, false, bF1, (short)0, acc01, false, false);
            acc10 = __builtin_amdgcn_wmma_f32_16x16x4_f32(false, aF1, false, bF0, (short)0, acc10, false, false);
            acc11 = __builtin_amdgcn_wmma_f32_16x16x4_f32(false, aF1, false, bF1, (short)0, acc11, false, false);
        }
#else
        const int off = hi ? 8 : 0;
        for (int k = 0; k < K; k += 32) {
            const v16h aF0 = load_f16_frag(A0 + off + k);
            const v16h aF1 = load_f16_frag(A1 + off + k);
            const v16h bF0 = load_f16_frag(B0 + off + k);
            const v16h bF1 = load_f16_frag(B1 + off + k);
            acc00 = __builtin_amdgcn_wmma_f32_16x16x32_f16(false, aF0, false, bF0, (short)0, acc00, false, false);
            acc01 = __builtin_amdgcn_wmma_f32_16x16x32_f16(false, aF0, false, bF1, (short)0, acc01, false, false);
            acc10 = __builtin_amdgcn_wmma_f32_16x16x32_f16(false, aF1, false, bF0, (short)0, acc10, false, false);
            acc11 = __builtin_amdgcn_wmma_f32_16x16x32_f16(false, aF1, false, bF1, (short)0, acc11, false, false);
        }
#endif
    }

    if (relu) {
#pragma unroll
        for (int r = 0; r < 8; ++r) {
            acc00[r] = fmaxf(acc00[r], 0.0f);
            acc01[r] = fmaxf(acc01[r], 0.0f);
            acc10[r] = fmaxf(acc10[r], 0.0f);
            acc11[r] = fmaxf(acc11[r], 0.0f);
        }
    }

    if (c0 < N) {
#pragma unroll
        for (int r = 0; r < 8; ++r) {
            const int m = m0 + r + (hi ? 8 : 0);
            C[(size_t)m * N + c0]        = acc00[r];
            C[(size_t)(m + 16) * N + c0] = acc10[r];
        }
    }
    if (c1 < N) {
#pragma unroll
        for (int r = 0; r < 8; ++r) {
            const int m = m0 + r + (hi ? 8 : 0);
            C[(size_t)m * N + c1]        = acc01[r];
            C[(size_t)(m + 16) * N + c1] = acc11[r];
        }
    }
}

// ---------------------------------------------------------------------------
// Launch: entropy(x) -> gemm1 -> entropy(h1) -> gemm2 -> entropy(h2) -> gemm3
// Workspace: h1 (8192*1024 f32) | h2 (8192*512 f32) | acc[3] f32  (~50.4 MB)
// ---------------------------------------------------------------------------
extern "C" void kernel_launch(void* const* d_in, const int* in_sizes, int n_in,
                              void* d_out, int out_size, void* d_ws, size_t ws_size,
                              hipStream_t stream)
{
    const float* x  = (const float*)d_in[0];
    const float* W1 = (const float*)d_in[1];
    const float* b1 = (const float*)d_in[2];
    const float* W2 = (const float*)d_in[3];
    const float* b2 = (const float*)d_in[4];
    const float* W3 = (const float*)d_in[5];
    const float* b3 = (const float*)d_in[6];
    float* out = (float*)d_out;

    float* h1  = (float*)d_ws;                              // 8192*1024
    float* h2  = h1 + (size_t)BATCH * H1_F;                 // 8192*512
    float* acc = h2 + (size_t)BATCH * H2_F;                 // 3 floats

    const float inv_scale = 1.0f / (8.0f * (float)BATCH);   // H/MAX_ENTROPY avg'd over B

    zero_acc_kernel<<<1, 1, 0, stream>>>(acc);

    // layer 1
    entropy_kernel<<<BATCH, 256, 0, stream>>>(x, IN_F, inv_scale, acc + 0);
    {
        dim3 grid((BATCH + 63) / 64, (H1_F + 127) / 128);
        gemm_entropy_branch<<<grid, 256, 0, stream>>>(
            x, W1, b1, h1, acc + 0, BATCH, H1_F, IN_F, /*relu=*/1);
    }

    // layer 2
    entropy_kernel<<<BATCH, 256, 0, stream>>>(h1, H1_F, inv_scale, acc + 1);
    {
        dim3 grid((BATCH + 63) / 64, (H2_F + 127) / 128);
        gemm_entropy_branch<<<grid, 256, 0, stream>>>(
            h1, W2, b2, h2, acc + 1, BATCH, H2_F, H1_F, /*relu=*/1);
    }

    // layer 3
    entropy_kernel<<<BATCH, 256, 0, stream>>>(h2, H2_F, inv_scale, acc + 2);
    {
        dim3 grid((BATCH + 63) / 64, (OUT_F + 127) / 128);
        gemm_entropy_branch<<<grid, 256, 0, stream>>>(
            h2, W3, b3, out, acc + 2, BATCH, OUT_F, H2_F, /*relu=*/0);
    }
}
